// BiSTSSM_78829829750911
// MI455X (gfx1250) — compile-verified, matrix operand order checked by reading, not run on hardware
//
#include <hip/hip_runtime.h>
#include <math.h>

// Problem constants (from reference)
#define DI    384     // d_inner
#define HJ    5       // joints / H
#define LT    2048    // sequence length
#define DFULL 1920    // DI*HJ
#define RRANK 12
#define NST   16
#define CROWS 44      // R + 2N
#define BATCH 4

typedef __attribute__((ext_vector_type(16))) __bf16 v16bf;
typedef __attribute__((ext_vector_type(8)))  float  v8f;

// ---------------------------------------------------------------------------
// Kernel 1: x_dbl[b,k,l,c] = sum_d W[k,c,d] * xs[b,k,d,l]   (bf16 WMMA, f32 acc)
// xs[b,0,d,l] = x[b, d/5, l, d%5];  xs[b,1,d,l] = x[b, d/5, L-1-l, d%5]
//
// K iterated in PERMUTED order d' = j*384 + di (same permutation on A and B
// leaves the dot product invariant). Outer loop over j, inner over di with
// explicitly carried pointers -> constant-stride, immediate-offset loads and
// affine induction (no div/mod, no address rematerialization).
//
// Rows 44..47 of the M padding use a CLAMPED row index (no EXEC divergence);
// their garbage results are simply never stored.
//
// WMMA 16-bit operand layout (CDNA5 ISA): element e of v16bf maps to
// K-offset ko = (e<8 ? e : e+8) + (lane>=16 ? 8 : 0); hi-half folded into base.
// ---------------------------------------------------------------------------
__global__ __launch_bounds__(128)
void xdbl_gemm_kernel(const float* __restrict__ x,      // (4,384,2048,5)
                      const float* __restrict__ w,      // (2,44,1920)
                      float* __restrict__ xdbl)         // (4,2,2048,44)
{
    const int lane  = threadIdx.x & 31;
    const int wave  = threadIdx.x >> 5;
    const int chunk = blockIdx.x & 31;     // 32 chunks of 64 columns
    const int bk    = blockIdx.x >> 5;     // 0..7
    const int b     = bk >> 1, k = bk & 1;
    const int lhalf = lane & 15;
    const int hi    = lane >> 4;

    const int n = chunk * 64 + wave * 16 + lhalf;      // output column (L index of xs)
    const int l = (k == 0) ? n : (LT - 1 - n);         // source column of x
    const int m2 = (lhalf + 32 < CROWS) ? (lhalf + 32) : 0;   // clamped pad row

    v8f acc0 = {}, acc1 = {}, acc2 = {};

    const size_t BSTR = (size_t)LT * HJ;               // 10240 floats per di step
    const float* bbase = x + (size_t)b * DI * BSTR + (size_t)(hi * 8) * BSTR + (size_t)l * HJ;
    const float* abase = w + (size_t)k * CROWS * DFULL + (hi * 8) * HJ;

    for (int j = 0; j < HJ; ++j) {
        const float* bp  = bbase + j;
        const float* a0p = abase + (size_t)lhalf        * DFULL + j;
        const float* a1p = abase + (size_t)(lhalf + 16) * DFULL + j;
        const float* a2p = abase + (size_t)m2           * DFULL + j;
        for (int it = 0; it < DI / 32; ++it) {
            v16bf bm, a0, a1, a2;
            #pragma unroll
            for (int e = 0; e < 16; ++e) {
                const int eo = (e < 8) ? e : e + 8;    // compile-time K offset
                bm[e] = (__bf16)bp[eo * (LT * HJ)];
                a0[e] = (__bf16)a0p[eo * HJ];
                a1[e] = (__bf16)a1p[eo * HJ];
                a2[e] = (__bf16)a2p[eo * HJ];
            }
            acc0 = __builtin_amdgcn_wmma_f32_16x16x32_bf16(false, a0, false, bm, (short)0, acc0, false, false);
            acc1 = __builtin_amdgcn_wmma_f32_16x16x32_bf16(false, a1, false, bm, (short)0, acc1, false, false);
            acc2 = __builtin_amdgcn_wmma_f32_16x16x32_bf16(false, a2, false, bm, (short)0, acc2, false, false);
            bp  += 32 * BSTR;
            a0p += 32 * HJ;
            a1p += 32 * HJ;
            a2p += 32 * HJ;
        }
    }

    // D layout: lane half -> N=lhalf; VGPR r -> M = r + (hi?8:0).
    // Store transposed: xdbl[b,k,l,c] so the scan reads contiguous columns.
    float* ob = xdbl + ((size_t)bk * LT + n) * CROWS;
    #pragma unroll
    for (int r = 0; r < 8; ++r) {
        int m = r + (hi ? 8 : 0);
        ob[m]      = acc0[r];
        ob[m + 16] = acc1[r];
        if (m + 32 < CROWS) ob[m + 32] = acc2[r];
    }
}

// ---------------------------------------------------------------------------
// Kernel 2: selective scan, one lane per (channel, state).
// Wave = 2 channels x 16 states (245,760 lanes = 7,680 wave32 waves).
// Per step a channel group touches one contiguous 176B x_dbl column (streaming
// in t). Channel group reduces y via shfl_xor; lane n==0 atomically accumulates
// into ys[b,d,l_src] (two commutative f32 adds per location -> deterministic).
// ---------------------------------------------------------------------------
__global__ __launch_bounds__(256)
void scan_kernel(const float* __restrict__ x,      // (4,384,2048,5)
                 const float* __restrict__ xdbl,   // (4,2,2048,44)
                 const float* __restrict__ dtw,    // (2,1920,12)
                 const float* __restrict__ dtb,    // (2,1920)
                 const float* __restrict__ Alog,   // (2*1920,16)
                 float* __restrict__ ys)           // (4,1920,2048) zero-initialized
{
    const int lane = threadIdx.x & 31;
    const int wave = threadIdx.x >> 5;
    const int n    = lane & 15;         // state index
    const int ch   = lane >> 4;         // which of the 2 channels in this wave
    // grid: 8 (b,k) groups x 120 blocks; block = 8 waves = 16 channels
    const int bk     = blockIdx.x / 120;
    const int blk_in = blockIdx.x - bk * 120;
    const int b = bk >> 1, k = bk & 1;
    const int d = blk_in * 16 + wave * 2 + ch;      // 0..1919
    const int di = d / HJ, j = d - di * HJ;

    // Per-lane constants
    float wreg[RRANK];
    #pragma unroll
    for (int r = 0; r < RRANK; ++r) wreg[r] = dtw[((size_t)k * DFULL + d) * RRANK + r];
    const float bias = dtb[(size_t)k * DFULL + d];
    const float Ad   = -__expf(Alog[((size_t)k * DFULL + d) * NST + n]);

    const float* col = xdbl + (size_t)(b * 2 + k) * LT * CROWS;   // (2048, 44)
    const float* xb  = x + ((size_t)b * DI + di) * LT * HJ + j;   // stride HJ over l
    float* yb = ys + ((size_t)b * DFULL + d) * LT;

    float h = 0.f;
    for (int t = 0; t < LT; ++t) {
        const int lsrc = k ? (LT - 1 - t) : t;
        const float* c44 = col + (size_t)t * CROWS;   // contiguous 44-value column
        // delta = softplus(dtw . dts[:,t] + bias)  (dts uniform across wave)
        float s = bias;
        #pragma unroll
        for (int r = 0; r < RRANK; ++r) s = fmaf(wreg[r], c44[r], s);
        const float delta = fmaxf(s, 0.f) + log1pf(__expf(-fabsf(s)));
        const float u  = xb[(size_t)lsrc * HJ];
        const float bs = c44[RRANK + n];
        const float cs = c44[RRANK + NST + n];
        h = __expf(delta * Ad) * h + (delta * u) * bs;
        float y = h * cs;
        // reduce over the 16 states (xor masks < 16 stay within the channel group)
        y += __shfl_xor(y, 1, 32);
        y += __shfl_xor(y, 2, 32);
        y += __shfl_xor(y, 4, 32);
        y += __shfl_xor(y, 8, 32);
        if (n == 0) atomicAdd(&yb[lsrc], y);
    }
}

// ---------------------------------------------------------------------------
// Kernel 3: add Ds*x, LayerNorm over d_inner=384, write (B,L,H,DI) output.
// One 128-thread block per (b,l,j); each thread owns 3 channels.
// ys reuse across neighboring l hits the 192 MB L2 (whole ys fits).
// ---------------------------------------------------------------------------
__global__ __launch_bounds__(128)
void ln_kernel(const float* __restrict__ x,
               const float* __restrict__ ys,
               const float* __restrict__ Ds,     // (2*1920,)
               const float* __restrict__ lnw,
               const float* __restrict__ lnb,
               float* __restrict__ out)          // (4,2048,5,384)
{
    const int blk = blockIdx.x;
    const int j = blk % HJ;
    const int l = (blk / HJ) % LT;
    const int b = blk / (HJ * LT);
    const int tid = threadIdx.x;

    __shared__ float sdata[8];

    float vals[3];
    float lsum = 0.f, lsq = 0.f;
    #pragma unroll
    for (int i = 0; i < 3; ++i) {
        const int di = tid + i * 128;
        const int d  = di * HJ + j;
        float v = ys[((size_t)b * DFULL + d) * LT + l]
                + (Ds[d] + Ds[DFULL + d]) * x[(((size_t)b * DI + di) * LT + l) * HJ + j];
        vals[i] = v;
        lsum += v;
        lsq  += v * v;
    }
    #pragma unroll
    for (int off = 16; off > 0; off >>= 1) {
        lsum += __shfl_xor(lsum, off, 32);
        lsq  += __shfl_xor(lsq,  off, 32);
    }
    const int wv = tid >> 5, ln = tid & 31;
    if (ln == 0) { sdata[wv] = lsum; sdata[4 + wv] = lsq; }
    __syncthreads();
    const float tsum = sdata[0] + sdata[1] + sdata[2] + sdata[3];
    const float tsq  = sdata[4] + sdata[5] + sdata[6] + sdata[7];
    const float mu  = tsum * (1.0f / DI);
    const float var = tsq * (1.0f / DI) - mu * mu;
    const float inv = rsqrtf(var + 1e-5f);
    #pragma unroll
    for (int i = 0; i < 3; ++i) {
        const int di = tid + i * 128;
        out[(((size_t)b * LT + l) * HJ + j) * DI + di] = (vals[i] - mu) * inv * lnw[di] + lnb[di];
    }
}

// ---------------------------------------------------------------------------
extern "C" void kernel_launch(void* const* d_in, const int* in_sizes, int n_in,
                              void* d_out, int out_size, void* d_ws, size_t ws_size,
                              hipStream_t stream) {
    const float* x    = (const float*)d_in[0];
    const float* w    = (const float*)d_in[1];
    const float* dtw  = (const float*)d_in[2];
    const float* dtb  = (const float*)d_in[3];
    const float* Alog = (const float*)d_in[4];
    const float* Ds   = (const float*)d_in[5];
    const float* lnw  = (const float*)d_in[6];
    const float* lnb  = (const float*)d_in[7];
    float* out = (float*)d_out;

    float* xdbl = (float*)d_ws;                                    // 4*2*2048*44 f32 = 2.9 MB
    float* ys   = xdbl + (size_t)BATCH * 2 * LT * CROWS;           // 4*1920*2048 f32 = 63 MB

    // ys must start at zero every call (atomic accumulation of 2 directions)
    hipMemsetAsync(ys, 0, (size_t)BATCH * DFULL * LT * sizeof(float), stream);

    // 8 (b,k) x 32 column-chunks, 128 threads (4 waves, 16 cols each)
    xdbl_gemm_kernel<<<256, 128, 0, stream>>>(x, w, xdbl);

    // 8 (b,k) x 120 blocks, 256 threads (8 waves x 2 channels x 16 states)
    scan_kernel<<<8 * 120, 256, 0, stream>>>(x, xdbl, dtw, dtb, Alog, ys);

    // one block per (b,l,j)
    ln_kernel<<<BATCH * LT * HJ, 128, 0, stream>>>(x, ys, Ds, lnw, lnb, out);
}
